// LocalAttention_40973988004715
// MI455X (gfx1250) — matrix operand order repaired and checked by reading, not compiled, not run on hardware
//
#include <hip/hip_runtime.h>
#include <cstdint>

#define DEVINL __device__ __forceinline__

constexpr int BB  = 8;
constexpr int NN  = 2048;
constexpr int BN  = BB * NN;     // 16384 rows
constexpr int DM  = 256;
constexpr int DQK = 128;
constexpr int NK  = 16;
constexpr int DFF = 1024;

typedef __attribute__((ext_vector_type(16))) _Float16 v16h;
typedef __attribute__((ext_vector_type(8)))  _Float16 v8h;
typedef __attribute__((ext_vector_type(8)))  float    v8f;

DEVINL v8f wmma32(v16h a, v16h b, v8f c) {
  return __builtin_amdgcn_wmma_f32_16x16x32_f16(false, a, false, b, (short)0, c,
                                                false, false);
}

// A fragment (16xK row-major source), K-chunk of 32 at kbase.
DEVINL v16h frag_a(const _Float16* A, int lda, int lane, int kbase) {
  const int row = lane & 15;
  const int klo = (lane >> 4) << 3;
  const _Float16* p = A + (size_t)row * lda + kbase + klo;
  v8h lo = *(const v8h*)p;
  v8h hi = *(const v8h*)(p + 16);
  v16h r;
#pragma unroll
  for (int i = 0; i < 8; ++i) { r[i] = lo[i]; r[i + 8] = hi[i]; }
  return r;
}

// B fragment: BT is B transposed, row-major [N][K].
DEVINL v16h frag_b(const _Float16* BT, int ldb, int lane, int kbase) {
  const int col = lane & 15;
  const int kk  = kbase + ((lane >> 4) << 3);
  const _Float16* p = BT + (size_t)col * ldb + kk;
  v8h lo = *(const v8h*)p;
  v8h hi = *(const v8h*)(p + 16);
  v16h r;
#pragma unroll
  for (int i = 0; i < 8; ++i) { r[i] = lo[i]; r[i + 8] = hi[i]; }
  return r;
}

// ---------------- prep ----------------
__global__ __launch_bounds__(256) void k_cvt_f16(const float* a, _Float16* b, int n) {
  int i = blockIdx.x * 256 + threadIdx.x;
  if (i < n) b[i] = (_Float16)a[i];
}

// W [K][Nn] (row-major) -> WT [Nn][K] f16
__global__ __launch_bounds__(256) void k_trans_f16(const float* W, _Float16* WT, int K, int Nn) {
  int i = blockIdx.x * 256 + threadIdx.x;
  if (i < K * Nn) {
    int n = i / K, k = i % K;
    WT[i] = (_Float16)W[(size_t)k * Nn + n];
  }
}

// ---------------- QK projection (pre-norm): 4 N-tiles per wave, A reused ----------------
__global__ __launch_bounds__(32) void k_proj_qk(const _Float16* xh,
                                                const _Float16* WqT, const _Float16* WkT,
                                                const float* bq, const float* bk,
                                                float* qpre, float* kpre) {
  const int lane = threadIdx.x;
  const int mt = blockIdx.x;            // 0..BN/16-1
  const int ng = blockIdx.y;            // 0..1 (group of 4 n-tiles)
  const bool isk = blockIdx.z != 0;
  const _Float16* BT = isk ? WkT : WqT;
  const float* bias  = isk ? bk  : bq;
  float* outp        = isk ? kpre : qpre;
  const _Float16* A = xh + (size_t)mt * 16 * DM;
  v16h af[8];
#pragma unroll
  for (int kc = 0; kc < 8; ++kc) af[kc] = frag_a(A, DM, lane, kc * 32);
  const int col = lane & 15, rb = (lane >> 4) << 3;
#pragma unroll
  for (int s = 0; s < 4; ++s) {
    const int nt = ng * 4 + s;
    v8f c = {};
#pragma unroll
    for (int kc = 0; kc < 8; ++kc)
      c = wmma32(af[kc], frag_b(BT + (size_t)nt * 16 * DM, DM, lane, kc * 32), c);
    const float bb = bias[nt * 16 + col];
#pragma unroll
    for (int j = 0; j < 8; ++j)
      outp[(size_t)(mt * 16 + rb + j) * DQK + nt * 16 + col] = c[j] + bb;
  }
}

// ---------------- L2 normalize rows of length 128 ----------------
__global__ __launch_bounds__(128) void k_l2norm(float* pre, _Float16* outh) {
  __shared__ float red[128];
  const int r = blockIdx.x, t = threadIdx.x;
  float v = pre[(size_t)r * DQK + t];
  red[t] = v * v;
  __syncthreads();
#pragma unroll
  for (int s = 64; s > 0; s >>= 1) {
    if (t < s) red[t] += red[t + s];
    __syncthreads();
  }
  float n = fmaxf(sqrtf(red[0]), 1e-12f);
  float o = v / n;
  pre[(size_t)r * DQK + t] = o;
  outh[(size_t)r * DQK + t] = (_Float16)o;
}

// ---------------- KNN: streaming top-16 by inner product ----------------
__global__ __launch_bounds__(32) void k_knn(const _Float16* qnh, const _Float16* knh,
                                            int* idxout) {
  __shared__ float stile[16][16];
  __shared__ float bestv[16][16];
  __shared__ int   besti[16][16];
  const int lane = threadIdx.x;
  const int qt = blockIdx.x;            // 0..NN/16-1
  const int b  = blockIdx.y;            // 0..BB-1
  const _Float16* Q = qnh + ((size_t)b * NN + qt * 16) * DQK;
  v16h a0 = frag_a(Q, DQK, lane, 0);
  v16h a1 = frag_a(Q, DQK, lane, 32);
  v16h a2 = frag_a(Q, DQK, lane, 64);
  v16h a3 = frag_a(Q, DQK, lane, 96);
  if (lane < 16) {
#pragma unroll
    for (int j = 0; j < 16; ++j) { bestv[lane][j] = -3.0e38f; besti[lane][j] = 0; }
  }
  __syncthreads();
  for (int kt = 0; kt < NN / 16; ++kt) {
    const _Float16* K = knh + ((size_t)b * NN + kt * 16) * DQK;
    v8f c = {};
    c = wmma32(a0, frag_b(K, DQK, lane, 0),  c);
    c = wmma32(a1, frag_b(K, DQK, lane, 32), c);
    c = wmma32(a2, frag_b(K, DQK, lane, 64), c);
    c = wmma32(a3, frag_b(K, DQK, lane, 96), c);
    const int col = lane & 15, rb = (lane >> 4) << 3;
#pragma unroll
    for (int j = 0; j < 8; ++j) stile[rb + j][col] = c[j];
    __syncthreads();
    if (lane < 16) {
      float worst = bestv[lane][15];
#pragma unroll 1
      for (int j = 0; j < 16; ++j) {
        float s = stile[lane][j];
        if (s > worst) {
          int p = 15;
          while (p > 0 && bestv[lane][p - 1] < s) {
            bestv[lane][p] = bestv[lane][p - 1];
            besti[lane][p] = besti[lane][p - 1];
            --p;
          }
          bestv[lane][p] = s;
          besti[lane][p] = kt * 16 + j;
          worst = bestv[lane][15];
        }
      }
    }
    __syncthreads();
  }
  if (lane < 16) {
#pragma unroll
    for (int j = 0; j < 16; ++j)
      idxout[((size_t)b * NN + qt * 16 + lane) * NK + j] = besti[lane][j];
  }
}

// ---------------- gather (async->LDS) + LN + V proj (WMMA) + softmax attention ----------------
__global__ __launch_bounds__(256) void k_attn(const float* x, const int* knnidx,
                                              const float* qn, const float* kn,
                                              const _Float16* WvT, const float* bvb,
                                              const float* g1, const float* b1,
                                              _Float16* oh) {
  __shared__ int nidx[16];
  __shared__ float vsm[16][DM];          // async-gathered x, then V results
  __shared__ _Float16 xvh[16][DM];       // layernormed gathered rows, f16
  __shared__ float rsum[16][16], rsum2[16][16];
  __shared__ float mean_s[16], rstd_s[16];
  __shared__ float logit[8][16], attnw[8][16];

  const int t = threadIdx.x, lane = t & 31, w = t >> 5;
  const int n = blockIdx.x;              // 0..BN-1
  const int b = n / NN;
  if (t < 16) nidx[t] = knnidx[(size_t)n * NK + t];
  __syncthreads();

  // gather 16 neighbor rows straight into LDS via async DMA (no VGPR round trip)
  const int r = t >> 4, sg = t & 15;
  const float* xr = x + ((size_t)b * NN + nidx[r]) * DM + sg * 16;
  unsigned lbase = (unsigned)(uintptr_t)&vsm[r][sg * 16];
#pragma unroll
  for (int i = 0; i < 4; ++i) {
    unsigned la = lbase + i * 16;
    const float* ga = xr + i * 4;
    asm volatile("global_load_async_to_lds_b128 %0, %1, off"
                 :: "v"(la), "v"(ga) : "memory");
  }
  asm volatile("s_wait_asynccnt 0" ::: "memory");
  __syncthreads();

  // LN partial sums (16 threads per row) reading from LDS
  float s1 = 0.f, s2 = 0.f;
#pragma unroll
  for (int i = 0; i < 16; ++i) {
    float v = vsm[r][sg * 16 + i];
    s1 += v; s2 += v * v;
  }
  rsum[r][sg] = s1; rsum2[r][sg] = s2;
  __syncthreads();
  if (t < 16) {
    float a = 0.f, q = 0.f;
#pragma unroll
    for (int i = 0; i < 16; ++i) { a += rsum[t][i]; q += rsum2[t][i]; }
    float m = a * (1.0f / DM);
    mean_s[t] = m;
    rstd_s[t] = rsqrtf(q * (1.0f / DM) - m * m + 1e-5f);
  }
  __syncthreads();
  {
    const float m = mean_s[r], rs = rstd_s[r];
#pragma unroll
    for (int i = 0; i < 16; ++i) {
      int c = sg * 16 + i;
      xvh[r][c] = (_Float16)((vsm[r][c] - m) * rs * g1[c] + b1[c]);
    }
  }
  __syncthreads();

  // V projection: each wave computes 2 column tiles of v = LN(x_v) @ Wv
  v16h af[8];
#pragma unroll
  for (int kc = 0; kc < 8; ++kc) af[kc] = frag_a(&xvh[0][0], DM, lane, kc * 32);
  v8f cc[2];
#pragma unroll
  for (int s = 0; s < 2; ++s) {
    const int nt = w + s * 8;
    v8f c = {};
#pragma unroll
    for (int kc = 0; kc < 8; ++kc)
      c = wmma32(af[kc], frag_b(WvT + (size_t)nt * 16 * DM, DM, lane, kc * 32), c);
    cc[s] = c;
  }
#pragma unroll
  for (int s = 0; s < 2; ++s) {
    const int nt = w + s * 8;
    const int col = lane & 15, rb = (lane >> 4) << 3;
    const float bias = bvb[nt * 16 + col];
#pragma unroll
    for (int j = 0; j < 8; ++j) vsm[rb + j][nt * 16 + col] = cc[s][j] + bias;
  }
  __syncthreads();

  // logits: (h,k) pairs, 16-wide head dot products, scale 1/sqrt(16)
  if (t < 128) {
    const int h = t >> 4, k = t & 15;
    const float* qp = qn + (size_t)n * DQK + h * 16;
    const float* kp = kn + ((size_t)b * NN + nidx[k]) * DQK + h * 16;
    float acc = 0.f;
#pragma unroll
    for (int d = 0; d < 16; ++d) acc += qp[d] * kp[d];
    logit[h][k] = acc * 0.25f;
  }
  __syncthreads();
  if (t < 8) {
    float mx = -3.0e38f;
#pragma unroll
    for (int k = 0; k < 16; ++k) mx = fmaxf(mx, logit[t][k]);
    float ssum = 0.f;
#pragma unroll
    for (int k = 0; k < 16; ++k) {
      float ev = expf(logit[t][k] - mx);
      attnw[t][k] = ev; ssum += ev;
    }
    float inv = 1.0f / ssum;
#pragma unroll
    for (int k = 0; k < 16; ++k) attnw[t][k] *= inv;
  }
  __syncthreads();

  // o[e] = sum_k attn[h(e)][k] * v[k][e]
  {
    const int e = t, h = e >> 5;   // dv = 32
    float acc = 0.f;
#pragma unroll
    for (int k = 0; k < 16; ++k) acc += attnw[h][k] * vsm[k][e];
    oh[(size_t)n * DM + e] = (_Float16)acc;
  }
}

// ---------------- output projection + residual: 4 N-tiles per wave ----------------
__global__ __launch_bounds__(32) void k_oproj(const _Float16* oh, const _Float16* WoT,
                                              const float* bo, const float* x,
                                              const float* rw, float* x1) {
  const int lane = threadIdx.x;
  const int mt = blockIdx.x, ng = blockIdx.y;   // ng 0..3
  const _Float16* A = oh + (size_t)mt * 16 * DM;
  v16h af[8];
#pragma unroll
  for (int kc = 0; kc < 8; ++kc) af[kc] = frag_a(A, DM, lane, kc * 32);
  const int col = lane & 15, rb = (lane >> 4) << 3;
  const float rwv = rw[0];
#pragma unroll
  for (int s = 0; s < 4; ++s) {
    const int nt = ng * 4 + s;
    v8f c = {};
#pragma unroll
    for (int kc = 0; kc < 8; ++kc)
      c = wmma32(af[kc], frag_b(WoT + (size_t)nt * 16 * DM, DM, lane, kc * 32), c);
    const float bb = bo[nt * 16 + col];
#pragma unroll
    for (int j = 0; j < 8; ++j) {
      size_t off = (size_t)(mt * 16 + rb + j) * DM + nt * 16 + col;
      x1[off] = x[off] + rwv * (c[j] + bb);
    }
  }
}

// ---------------- row LayerNorm over 256, writes f16 ----------------
__global__ __launch_bounds__(256) void k_ln(const float* x1, const float* gf,
                                            const float* bf, _Float16* xnh) {
  __shared__ float red[256], red2[256];
  const int r = blockIdx.x, t = threadIdx.x;
  float v = x1[(size_t)r * DM + t];
  red[t] = v; red2[t] = v * v;
  __syncthreads();
#pragma unroll
  for (int s = 128; s > 0; s >>= 1) {
    if (t < s) { red[t] += red[t + s]; red2[t] += red2[t + s]; }
    __syncthreads();
  }
  float m = red[0] * (1.0f / DM);
  float rs = rsqrtf(red2[0] * (1.0f / DM) - m * m + 1e-5f);
  xnh[(size_t)r * DM + t] = (_Float16)((v - m) * rs * gf[t] + bf[t]);
}

// ---------------- FFN1: GEMM + bias + exact GELU, 4 N-tiles per wave ----------------
__global__ __launch_bounds__(32) void k_ffn1(const _Float16* xnh, const _Float16* W1T,
                                             const float* b1f, _Float16* hh) {
  const int lane = threadIdx.x;
  const int mt = blockIdx.x, ng = blockIdx.y;   // ng 0..15
  const _Float16* A = xnh + (size_t)mt * 16 * DM;
  v16h af[8];
#pragma unroll
  for (int kc = 0; kc < 8; ++kc) af[kc] = frag_a(A, DM, lane, kc * 32);
  const int col = lane & 15, rb = (lane >> 4) << 3;
#pragma unroll
  for (int s = 0; s < 4; ++s) {
    const int nt = ng * 4 + s;
    v8f c = {};
#pragma unroll
    for (int kc = 0; kc < 8; ++kc)
      c = wmma32(af[kc], frag_b(W1T + (size_t)nt * 16 * DM, DM, lane, kc * 32), c);
    const float bb = b1f[nt * 16 + col];
#pragma unroll
    for (int j = 0; j < 8; ++j) {
      float v = c[j] + bb;
      float g = 0.5f * v * (1.0f + erff(v * 0.70710678118654752f));
      hh[(size_t)(mt * 16 + rb + j) * DFF + nt * 16 + col] = (_Float16)g;
    }
  }
}

// ---------------- FFN2: K=1024 GEMM, A streamed once per chunk feeding 4 tiles ----------------
__global__ __launch_bounds__(32) void k_ffn2(const _Float16* hh, const _Float16* W2T,
                                             const float* b2f, const float* x1,
                                             const float* rw, float* out) {
  const int lane = threadIdx.x;
  const int mt = blockIdx.x, ng = blockIdx.y;   // ng 0..3
  const _Float16* A = hh + (size_t)mt * 16 * DFF;
  v8f c[4] = {{}, {}, {}, {}};
#pragma unroll 4
  for (int kc = 0; kc < DFF / 32; ++kc) {
    v16h a = frag_a(A, DFF, lane, kc * 32);
#pragma unroll
    for (int s = 0; s < 4; ++s)
      c[s] = wmma32(a, frag_b(W2T + (size_t)(ng * 4 + s) * 16 * DFF, DFF, lane, kc * 32),
                    c[s]);
  }
  const int col = lane & 15, rb = (lane >> 4) << 3;
  const float rwv = rw[0];
#pragma unroll
  for (int s = 0; s < 4; ++s) {
    const int nt = ng * 4 + s;
    const float bb = b2f[nt * 16 + col];
#pragma unroll
    for (int j = 0; j < 8; ++j) {
      size_t off = (size_t)(mt * 16 + rb + j) * DM + nt * 16 + col;
      out[off] = x1[off] + rwv * (c[s][j] + bb);
    }
  }
}

extern "C" void kernel_launch(void* const* d_in, const int* in_sizes, int n_in,
                              void* d_out, int out_size, void* d_ws, size_t ws_size,
                              hipStream_t stream) {
  const float* x   = (const float*)d_in[0];
  const float* Wq  = (const float*)d_in[1];
  const float* bq  = (const float*)d_in[2];
  const float* Wk  = (const float*)d_in[3];
  const float* bk  = (const float*)d_in[4];
  const float* Wv  = (const float*)d_in[5];
  const float* bv  = (const float*)d_in[6];
  const float* Wo  = (const float*)d_in[7];
  const float* bo  = (const float*)d_in[8];
  const float* g1  = (const float*)d_in[9];
  const float* b1  = (const float*)d_in[10];
  const float* gf  = (const float*)d_in[11];
  const float* bf  = (const float*)d_in[12];
  const float* W1  = (const float*)d_in[13];
  const float* b1f = (const float*)d_in[14];
  const float* W2  = (const float*)d_in[15];
  const float* b2f = (const float*)d_in[16];
  const float* rw  = (const float*)d_in[17];
  float* out = (float*)d_out;

  char* p = (char*)d_ws;
  auto carve = [&](size_t bytes) -> char* {
    char* r = p;
    p += (bytes + 255) & ~(size_t)255;
    return r;
  };
  _Float16* xh   = (_Float16*)carve((size_t)BN * DM * 2);
  _Float16* WqT  = (_Float16*)carve((size_t)DQK * DM * 2);
  _Float16* WkT  = (_Float16*)carve((size_t)DQK * DM * 2);
  _Float16* WvT  = (_Float16*)carve((size_t)DM * DM * 2);
  _Float16* WoT  = (_Float16*)carve((size_t)DM * DM * 2);
  _Float16* W1T  = (_Float16*)carve((size_t)DFF * DM * 2);
  _Float16* W2T  = (_Float16*)carve((size_t)DM * DFF * 2);
  float*    qn   = (float*)carve((size_t)BN * DQK * 4);    // pre-norm, normalized in place
  float*    kn   = (float*)carve((size_t)BN * DQK * 4);
  _Float16* qnh  = (_Float16*)carve((size_t)BN * DQK * 2);
  _Float16* knh  = (_Float16*)carve((size_t)BN * DQK * 2);
  int*      kidx = (int*)carve((size_t)BN * NK * 4);
  _Float16* oh   = (_Float16*)carve((size_t)BN * DM * 2);
  float*    x1   = (float*)carve((size_t)BN * DM * 4);
  _Float16* xnh  = (_Float16*)carve((size_t)BN * DM * 2);
  _Float16* hh   = (_Float16*)carve((size_t)BN * DFF * 2);

  // prep: f16 activations + transposed f16 weights
  k_cvt_f16<<<(BN * DM + 255) / 256, 256, 0, stream>>>(x, xh, BN * DM);
  k_trans_f16<<<(DM * DQK + 255) / 256, 256, 0, stream>>>(Wq, WqT, DM, DQK);
  k_trans_f16<<<(DM * DQK + 255) / 256, 256, 0, stream>>>(Wk, WkT, DM, DQK);
  k_trans_f16<<<(DM * DM + 255) / 256, 256, 0, stream>>>(Wv, WvT, DM, DM);
  k_trans_f16<<<(DM * DM + 255) / 256, 256, 0, stream>>>(Wo, WoT, DM, DM);
  k_trans_f16<<<(DM * DFF + 255) / 256, 256, 0, stream>>>(W1, W1T, DM, DFF);
  k_trans_f16<<<(DM * DFF + 255) / 256, 256, 0, stream>>>(W2, W2T, DFF, DM);

  // QK projection + L2 normalize
  k_proj_qk<<<dim3(BN / 16, 2, 2), 32, 0, stream>>>(xh, WqT, WkT, bq, bk, qn, kn);
  k_l2norm<<<BN, 128, 0, stream>>>(qn, qnh);
  k_l2norm<<<BN, 128, 0, stream>>>(kn, knh);

  // KNN top-16 by inner product (== smallest Euclidean distance on unit vectors)
  k_knn<<<dim3(NN / 16, BB), 32, 0, stream>>>(qnh, knh, kidx);

  // gather + LN + V projection + windowed attention
  k_attn<<<BN, 256, 0, stream>>>(x, kidx, qn, kn, WvT, bv, g1, b1, oh);

  // output projection + residual
  k_oproj<<<dim3(BN / 16, 4), 32, 0, stream>>>(oh, WoT, bo, x, rw, x1);

  // FFN
  k_ln<<<BN, 256, 0, stream>>>(x1, gf, bf, xnh);
  k_ffn1<<<dim3(BN / 16, 16), 32, 0, stream>>>(xnh, W1T, b1f, hh);
  k_ffn2<<<dim3(BN / 16, 4), 32, 0, stream>>>(hh, W2T, b2f, x1, rw, out);
}
